// LSTMModel_77773267796141
// MI455X (gfx1250) — matrix operand order repaired and checked by reading, not compiled, not run on hardware
//
#include <hip/hip_runtime.h>

#define VOCAB  32000
#define EMBED  256
#define HIDDEN 1024
#define BATCH  64
#define SEQT   256

typedef __attribute__((ext_vector_type(16))) __bf16 v16bf;
typedef __attribute__((ext_vector_type(8)))  __bf16 bf16x8;
typedef __attribute__((ext_vector_type(8)))  float  v8f;

#define LDS_STRIDE 40   // 32 used + 8 pad; row pitch = 80B (16B aligned, spreads banks)
#define A_BUF (64  * LDS_STRIDE)
#define B_BUF (64  * LDS_STRIDE)
#define BFC_BUF (128 * LDS_STRIDE)

__device__ __forceinline__ float fsigmoid(float x) {
    return __builtin_amdgcn_rcpf(1.0f + __expf(-x));
}
__device__ __forceinline__ float ftanh(float x) {
    float e = __expf(2.0f * x);
    return (e - 1.0f) * __builtin_amdgcn_rcpf(e + 1.0f);
}

// Convert 16 f32 (4x float4) -> 16 bf16, store as two 16B LDS writes.
__device__ __forceinline__ void cvt_store16(__bf16* d, float4 a, float4 b, float4 c, float4 e) {
    bf16x8 v0 = { (__bf16)a.x, (__bf16)a.y, (__bf16)a.z, (__bf16)a.w,
                  (__bf16)b.x, (__bf16)b.y, (__bf16)b.z, (__bf16)b.w };
    bf16x8 v1 = { (__bf16)c.x, (__bf16)c.y, (__bf16)c.z, (__bf16)c.w,
                  (__bf16)e.x, (__bf16)e.y, (__bf16)e.z, (__bf16)e.w };
    *(bf16x8*)(d)     = v0;
    *(bf16x8*)(d + 8) = v1;
}

__device__ __forceinline__ v16bf cat8(bf16x8 r0, bf16x8 r1) {
    return __builtin_shufflevector(r0, r1, 0,1,2,3,4,5,6,7,8,9,10,11,12,13,14,15);
}

// A fragment (16x32 bf16, MxK). Layout: lanes 0-15: M=lane, K=0..15;
// lanes 16-31: M=lane-16, K=16..31 (matches B fragment K order).
__device__ __forceinline__ v16bf load_a_frag(const __bf16* lds, int row_base) {
    const int lane = threadIdx.x & 31;
    const __bf16* p = lds + (row_base + (lane & 15)) * LDS_STRIDE;
    const int lo = (lane < 16);
    bf16x8 r0 = *(const bf16x8*)(p + (lo ? 0 : 16));
    bf16x8 r1 = *(const bf16x8*)(p + (lo ? 8 : 24));
    return cat8(r0, r1);
}

// B fragment (32x16 bf16, KxN), tile staged transposed in LDS as [n][k].
// Layout used here (same k order as A): lanes 0-15: N=lane, K=0..15; lanes 16-31: N=lane-16, K=16..31.
__device__ __forceinline__ v16bf load_b_frag(const __bf16* lds, int col_base) {
    const int lane = threadIdx.x & 31;
    const __bf16* p = lds + (col_base + (lane & 15)) * LDS_STRIDE + ((lane < 16) ? 0 : 16);
    bf16x8 r0 = *(const bf16x8*)(p);
    bf16x8 r1 = *(const bf16x8*)(p + 8);
    return cat8(r0, r1);
}

// ---------------- staged-register helpers (software pipeline) ----------------

struct StageH  { uint4 a0, a1; float4 w0, w1, w2, w3; };            // A: bf16 copy, B: f32
struct StageE  { float4 e0, e1, e2, e3, w0, w1, w2, w3; };          // A: f32, B: f32
struct StageFC { uint4 a0, a1; float4 w0, w1, w2, w3, w4, w5, w6, w7; };

__device__ __forceinline__ StageH ld_stage_h(const __bf16* h_read, const float* whh_row,
                                             int kc, int srow, int shalf) {
    StageH s;
    const uint4* a = (const uint4*)(h_read + srow * HIDDEN + kc * 32 + shalf * 16);
    s.a0 = a[0]; s.a1 = a[1];
    const float4* w = (const float4*)(whh_row + kc * 32 + shalf * 16);
    s.w0 = w[0]; s.w1 = w[1]; s.w2 = w[2]; s.w3 = w[3];
    return s;
}
__device__ __forceinline__ void st_stage_h(const StageH& s, __bf16* la, __bf16* lb,
                                           int srow, int shalf) {
    uint4* dst = (uint4*)(la + srow * LDS_STRIDE + shalf * 16);
    dst[0] = s.a0; dst[1] = s.a1;
    cvt_store16(lb + srow * LDS_STRIDE + shalf * 16, s.w0, s.w1, s.w2, s.w3);
}

__device__ __forceinline__ StageE ld_stage_e(const float* emb_row, const float* wih_row,
                                             int kc, int shalf) {
    StageE s;
    const float4* e = (const float4*)(emb_row + kc * 32 + shalf * 16);
    s.e0 = e[0]; s.e1 = e[1]; s.e2 = e[2]; s.e3 = e[3];
    const float4* w = (const float4*)(wih_row + kc * 32 + shalf * 16);
    s.w0 = w[0]; s.w1 = w[1]; s.w2 = w[2]; s.w3 = w[3];
    return s;
}
__device__ __forceinline__ void st_stage_e(const StageE& s, __bf16* la, __bf16* lb,
                                           int srow, int shalf) {
    cvt_store16(la + srow * LDS_STRIDE + shalf * 16, s.e0, s.e1, s.e2, s.e3);
    cvt_store16(lb + srow * LDS_STRIDE + shalf * 16, s.w0, s.w1, s.w2, s.w3);
}

__device__ __forceinline__ void compute4(const __bf16* la, const __bf16* lb,
                                         int wave, v8f acc[4]) {
    v16bf a = load_a_frag(la, wave * 16);
#pragma unroll
    for (int g = 0; g < 4; ++g) {
        v16bf b = load_b_frag(lb, g * 16);
        acc[g] = __builtin_amdgcn_wmma_f32_16x16x32_bf16(
            false, a, false, b, (short)0, acc[g], false, false);
    }
}

// -----------------------------------------------------------------------------

__global__ void init_state_kernel(float* __restrict__ c_state, unsigned short* __restrict__ h0) {
    int i = blockIdx.x * blockDim.x + threadIdx.x;
    if (i < BATCH * HIDDEN) {
        c_state[i] = 0.0f;
        h0[i] = 0;            // bf16 +0.0
    }
}

// One fused LSTM time step:
//   z[:, g*H+j] = emb(x[:,t]) . W_ih^T + h . W_hh^T + (b_ih + b_hh)
//   i,f,g,o -> c' = f*c + i*g ; h' = o*tanh(c')
// Grid: 64 WGs (16 hidden units each) x 128 threads (4 waves; wave = 16-row M-tile).
// Double-buffered LDS, one barrier per K-chunk; K = 1024 (W_hh) + 256 (W_ih).
__global__ __launch_bounds__(128) void lstm_step_kernel(
    const int*   __restrict__ x,
    const float* __restrict__ embed_table,
    const float* __restrict__ W_ih,
    const float* __restrict__ W_hh,
    const float* __restrict__ b_ih,
    const float* __restrict__ b_hh,
    const unsigned short* __restrict__ h_read_u16,   // bf16 [64][1024]
    unsigned short*       __restrict__ h_write_u16,  // bf16 [64][1024]
    float* __restrict__ c_state,                     // f32  [64][1024]
    int t)
{
    __shared__ __align__(16) __bf16 lds_a[2 * A_BUF];
    __shared__ __align__(16) __bf16 lds_b[2 * B_BUF];

    const __bf16* h_read  = (const __bf16*)h_read_u16;
    __bf16*       h_write = (__bf16*)h_write_u16;

    const int tid  = threadIdx.x;
    const int wave = tid >> 5;
    const int lane = tid & 31;
    const int j0   = blockIdx.x * 16;

    const int srow  = tid >> 1;          // staging row 0..63 (2 threads/row)
    const int shalf = tid & 1;

    const int gate    = srow >> 4;
    const int nrow    = gate * HIDDEN + j0 + (srow & 15);    // global W row for B staging
    const int emb_idx = x[srow * SEQT + t];
    const float* emb_row = embed_table + (size_t)emb_idx * EMBED;
    const float* whh_row = W_hh + (size_t)nrow * HIDDEN;
    const float* wih_row = W_ih + (size_t)nrow * EMBED;

    v8f acc[4];
    const v8f vzero = {0.f,0.f,0.f,0.f,0.f,0.f,0.f,0.f};
#pragma unroll
    for (int g = 0; g < 4; ++g) acc[g] = vzero;

    // ---- prologue: stage chunk 0 (h / W_hh phase) ----
    {
        StageH s0 = ld_stage_h(h_read, whh_row, 0, srow, shalf);
        st_stage_h(s0, lds_a, lds_b, srow, shalf);
    }
    __syncthreads();

    // ---- phase 1: chunks 0..31 feed h.W_hh^T; pipeline depth 1 ----
    for (int kc = 0; kc < 31; ++kc) {
        StageH s = ld_stage_h(h_read, whh_row, kc + 1, srow, shalf);
        compute4(lds_a + (kc & 1) * A_BUF, lds_b + (kc & 1) * B_BUF, wave, acc);
        st_stage_h(s, lds_a + ((kc + 1) & 1) * A_BUF, lds_b + ((kc + 1) & 1) * B_BUF, srow, shalf);
        __syncthreads();
    }
    // ---- transition: compute chunk 31, stage first emb/W_ih chunk (global chunk 32) ----
    {
        StageE s = ld_stage_e(emb_row, wih_row, 0, shalf);
        compute4(lds_a + A_BUF, lds_b + B_BUF, wave, acc);         // 31 & 1 == 1
        st_stage_e(s, lds_a, lds_b, srow, shalf);                  // 32 & 1 == 0
        __syncthreads();
    }
    // ---- phase 2: chunks 32..39 feed emb.W_ih^T ----
    for (int kc = 32; kc < 39; ++kc) {
        StageE s = ld_stage_e(emb_row, wih_row, kc + 1 - 32, shalf);
        compute4(lds_a + (kc & 1) * A_BUF, lds_b + (kc & 1) * B_BUF, wave, acc);
        st_stage_e(s, lds_a + ((kc + 1) & 1) * A_BUF, lds_b + ((kc + 1) & 1) * B_BUF, srow, shalf);
        __syncthreads();
    }
    // ---- epilogue compute: chunk 39 ----
    compute4(lds_a + A_BUF, lds_b + B_BUF, wave, acc);             // 39 & 1 == 1

    // ---- LSTM cell update: per-lane on the C-matrix layout ----
    // VGPR r: lanes 0-15 -> row r, lanes 16-31 -> row r+8; column = lane&15.
    const int jl  = lane & 15;
    const int col = j0 + jl;
    float bias[4];
#pragma unroll
    for (int g = 0; g < 4; ++g) bias[g] = b_ih[g * HIDDEN + col] + b_hh[g * HIDDEN + col];

    const int rbase = wave * 16 + ((lane >> 4) << 3);
#pragma unroll
    for (int r = 0; r < 8; ++r) {
        const int b  = rbase + r;
        const float zi = acc[0][r] + bias[0];
        const float zf = acc[1][r] + bias[1];
        const float zg = acc[2][r] + bias[2];
        const float zo = acc[3][r] + bias[3];
        const float ig = fsigmoid(zi);
        const float fg = fsigmoid(zf);
        const float gg = ftanh(zg);
        const float og = fsigmoid(zo);
        const float cn = fg * c_state[b * HIDDEN + col] + ig * gg;
        c_state[b * HIDDEN + col] = cn;
        h_write[b * HIDDEN + col] = (__bf16)(og * ftanh(cn));
    }
}

// out[b, v] = h_last . fc_W^T + fc_b.  M=64, N=32000, K=1024.
// Grid: 250 WGs x 128 threads; wave = M-tile, 8 N-tiles per wave. Double-buffered.
__global__ __launch_bounds__(128) void fc_kernel(
    const unsigned short* __restrict__ h_u16,   // bf16 [64][1024]
    const float* __restrict__ fc_W,             // [32000][1024]
    const float* __restrict__ fc_b,
    float* __restrict__ out)                    // [64][32000]
{
    __shared__ __align__(16) __bf16 lds_a[2 * A_BUF];
    __shared__ __align__(16) __bf16 lds_b[2 * BFC_BUF];

    const __bf16* h = (const __bf16*)h_u16;
    const int tid  = threadIdx.x;
    const int wave = tid >> 5;
    const int lane = tid & 31;
    const int v0   = blockIdx.x * 128;

    const int srow  = tid >> 1;
    const int shalf = tid & 1;
    const float* wrow = fc_W + (size_t)(v0 + tid) * HIDDEN;

    v8f acc[8];
    const v8f vzero = {0.f,0.f,0.f,0.f,0.f,0.f,0.f,0.f};
#pragma unroll
    for (int n = 0; n < 8; ++n) acc[n] = vzero;

    // staging: A 2x uint4 (bf16 copy), B one full 32-f32 row chunk per thread.
    auto ld_fc = [&](int kc) {
        StageFC s;
        const uint4* a = (const uint4*)(h + srow * HIDDEN + kc * 32 + shalf * 16);
        s.a0 = a[0]; s.a1 = a[1];
        const float4* w = (const float4*)(wrow + kc * 32);
        s.w0 = w[0]; s.w1 = w[1]; s.w2 = w[2]; s.w3 = w[3];
        s.w4 = w[4]; s.w5 = w[5]; s.w6 = w[6]; s.w7 = w[7];
        return s;
    };
    auto st_fc = [&](const StageFC& s, __bf16* la, __bf16* lb) {
        uint4* dst = (uint4*)(la + srow * LDS_STRIDE + shalf * 16);
        dst[0] = s.a0; dst[1] = s.a1;
        cvt_store16(lb + tid * LDS_STRIDE,      s.w0, s.w1, s.w2, s.w3);
        cvt_store16(lb + tid * LDS_STRIDE + 16, s.w4, s.w5, s.w6, s.w7);
    };
    auto compute8 = [&](const __bf16* la, const __bf16* lb) {
        v16bf a = load_a_frag(la, wave * 16);
#pragma unroll
        for (int nt = 0; nt < 8; ++nt) {
            v16bf b = load_b_frag(lb, nt * 16);
            acc[nt] = __builtin_amdgcn_wmma_f32_16x16x32_bf16(
                false, a, false, b, (short)0, acc[nt], false, false);
        }
    };

    {
        StageFC s0 = ld_fc(0);
        st_fc(s0, lds_a, lds_b);
    }
    __syncthreads();

    for (int kc = 0; kc < 31; ++kc) {
        // branchless clamped prefetch two chunks ahead (-> global_prefetch_b8)
        const int pk = (kc + 2 < 32) ? (kc + 2) : 31;
        __builtin_prefetch(wrow + pk * 32, 0, 3);
        StageFC s = ld_fc(kc + 1);
        compute8(lds_a + (kc & 1) * A_BUF, lds_b + (kc & 1) * BFC_BUF);
        st_fc(s, lds_a + ((kc + 1) & 1) * A_BUF, lds_b + ((kc + 1) & 1) * BFC_BUF);
        __syncthreads();
    }
    compute8(lds_a + A_BUF, lds_b + BFC_BUF);                      // 31 & 1 == 1

    const int jl    = lane & 15;
    const int rbase = wave * 16 + ((lane >> 4) << 3);
#pragma unroll
    for (int nt = 0; nt < 8; ++nt) {
        const int v  = v0 + nt * 16 + jl;
        const float bb = fc_b[v];
#pragma unroll
        for (int r = 0; r < 8; ++r) {
            out[(size_t)(rbase + r) * VOCAB + v] = acc[nt][r] + bb;
        }
    }
}

extern "C" void kernel_launch(void* const* d_in, const int* in_sizes, int n_in,
                              void* d_out, int out_size, void* d_ws, size_t ws_size,
                              hipStream_t stream) {
    (void)in_sizes; (void)n_in; (void)out_size; (void)ws_size;
    const int*   x           = (const int*)  d_in[0];
    const float* embed_table = (const float*)d_in[1];
    const float* W_ih        = (const float*)d_in[2];
    const float* W_hh        = (const float*)d_in[3];
    const float* b_ih        = (const float*)d_in[4];
    const float* b_hh        = (const float*)d_in[5];
    const float* fc_W        = (const float*)d_in[6];
    const float* fc_b        = (const float*)d_in[7];
    float*       out         = (float*)d_out;

    char* ws = (char*)d_ws;
    float*          c_state = (float*)ws;                                       // 256 KB
    unsigned short* h_buf0  = (unsigned short*)(ws + BATCH * HIDDEN * 4);       // 128 KB bf16
    unsigned short* h_buf1  = (unsigned short*)(ws + BATCH * HIDDEN * 4
                                                   + BATCH * HIDDEN * 2);       // 128 KB bf16

    init_state_kernel<<<(BATCH * HIDDEN + 255) / 256, 256, 0, stream>>>(c_state, h_buf0);

    for (int t = 0; t < SEQT; ++t) {
        unsigned short* hr = (t & 1) ? h_buf1 : h_buf0;
        unsigned short* hw = (t & 1) ? h_buf0 : h_buf1;
        lstm_step_kernel<<<HIDDEN / 16, 128, 0, stream>>>(
            x, embed_table, W_ih, W_hh, b_ih, b_hh, hr, hw, c_state, t);
    }
    // T=256 is even: final hidden state lands back in h_buf0.
    fc_kernel<<<VOCAB / 128, 128, 0, stream>>>(h_buf0, fc_W, fc_b, out);
}
